// _Expert_11871289606691
// MI455X (gfx1250) — matrix operand order repaired and checked by reading, not compiled, not run on hardware
//
#include <hip/hip_runtime.h>

// ---------------------------------------------------------------------------
// MoE grouped GEMM: out[t] = inp[t] @ W[e]^T + b[e], experts segmented by
// fwd_expert_count. fp32 in/out, bf16 V_WMMA_F32_16X16X32_BF16 compute.
// B tiles staged via gfx1250 GLOBAL_LOAD_ASYNC_TO_LDS in GVS mode (ASYNCcnt),
// f32->bf16 packing via single V_PERM_B32 per pair. A rows clamped (not
// guarded): out-of-count rows only feed C rows the epilogue never stores.
// ---------------------------------------------------------------------------

#define NUM_EXPERTS 64
#define DIN  512
#define DOUT 512
#define BM   128
#define BN   128
#define BK   32
#define KTILES (DIN / BK)      // 16
#define LDSS 40                // bf16 elements per LDS A row (32 + 8 pad)
#define BSTRIDE 36             // f32 elements per LDS B row (32 + 4 pad, 144B)
#define MAX_TILES 1088         // T/BM + E upper bound

typedef __bf16 bf16_t;
typedef __attribute__((ext_vector_type(16))) bf16_t v16bf;
typedef __attribute__((ext_vector_type(8)))  float  v8f;

struct U4 { unsigned x, y, z, w; };
struct U2 { unsigned x, y; };
struct F4 { float x, y, z, w; };
struct V16B { U4 lo, hi; };          // 32 bytes == one v16bf fragment
struct Tile { int e; int row0; int rows; int pad; };

// pack2(hi,lo): one v_perm_b32 -> {bf16(hi), bf16(lo)} (truncating f32->bf16)
__device__ __forceinline__ unsigned pack2(unsigned hi_bits, unsigned lo_bits) {
  return __builtin_amdgcn_perm(hi_bits, lo_bits, 0x07060302u);
}
__device__ __forceinline__ unsigned pack2f(float lo, float hi) {
  return pack2(__float_as_uint(hi), __float_as_uint(lo));
}

// gfx1250 async copy, GVS mode: mem = SGPR64 base + VGPR32 offset -> LDS.
// Per-lane vaddr is loop-invariant; K-advance rides the uniform SGPR base.
__device__ __forceinline__ void async_copy_b128_gvs(unsigned lds_off, unsigned voff,
                                                    unsigned long long sbase) {
  asm volatile("global_load_async_to_lds_b128 %0, %1, %2"
               :: "v"(lds_off), "v"(voff), "s"(sbase)
               : "memory");
}
__device__ __forceinline__ void wait_async0() {
#if __has_builtin(__builtin_amdgcn_s_wait_asynccnt)
  __builtin_amdgcn_s_wait_asynccnt(0);
#else
  asm volatile("s_wait_asynccnt 0x0" ::: "memory");
#endif
}

// ---------------------------------------------------------------------------
// Build tile table from int64 expert counts (single thread; 64 experts).
// ws layout: [0..15] int meta (meta[0]=numTiles), [16..] Tile entries.
// ---------------------------------------------------------------------------
__global__ void build_tiles_kernel(const long long* __restrict__ cnt,
                                   int* __restrict__ meta,
                                   Tile* __restrict__ tiles) {
  if (threadIdx.x != 0 || blockIdx.x != 0) return;
  int nt = 0;
  long long off = 0;
  for (int e = 0; e < NUM_EXPERTS; ++e) {
    long long c = cnt[e];
    int ntile = (int)((c + (BM - 1)) / BM);
    for (int i = 0; i < ntile && nt < MAX_TILES; ++i) {
      long long rem = c - (long long)i * BM;
      tiles[nt].e    = e;
      tiles[nt].row0 = (int)(off + (long long)i * BM);
      tiles[nt].rows = (int)(rem < BM ? rem : BM);
      tiles[nt].pad  = 0;
      ++nt;
    }
    off += c;
  }
  meta[0] = nt;
}

// ---------------------------------------------------------------------------
// Main GEMM kernel. grid = (DOUT/BN, MAX_TILES), block = 256 (8 waves).
// ---------------------------------------------------------------------------
__global__ __launch_bounds__(256) void moe_gemm_kernel(
    const float* __restrict__ inp, const float* __restrict__ weight,
    const float* __restrict__ bias, const int* __restrict__ meta,
    const Tile* __restrict__ tiles, float* __restrict__ out) {
  const int numTiles = meta[0];
  const int tIdx = blockIdx.y;
  if (tIdx >= numTiles) return;
  const Tile t = tiles[tIdx];
  const int n0 = blockIdx.x * BN;

  __shared__ __align__(16) unsigned short sA[2][BM * LDSS];     // bf16 A tiles
  __shared__ __align__(16) float          sBf[2][BN * BSTRIDE]; // f32 B tiles

  const int tid   = threadIdx.x;
  const int lane  = tid & 31;
  const int l15   = lane & 15;
  const int lhalf = lane >> 4;            // 0 or 1 (half-wave)
  const int wave  = tid >> 5;
  const int waveM = (wave & 1) * 64;      // 2 wave-rows of 64
  const int waveN = (wave >> 1) * 32;     // 4 wave-cols of 32

  const unsigned long long Bsad0 =
      (unsigned long long)(size_t)(weight + (size_t)t.e * (DOUT * DIN) + (size_t)n0 * DIN);

  // ---- loop-invariant per-thread staging state (4 float4 segments each) ----
  const float* aptr[4];     // clamped, always-in-bounds A row pointers
  unsigned     boffv[4];    // per-lane byte offsets for async B (GVS vaddr)
  unsigned     bdst0[4];    // LDS byte addresses of B dest, buffer 0
  unsigned     adst0[4];    // LDS element offsets of A dest, buffer 0
#pragma unroll
  for (int i = 0; i < 4; ++i) {
    const int seg = tid + 256 * i;
    const int r = seg >> 3;            // 8 float4 per 32-float row
    const int c = (seg & 7) << 2;      // float column
    const int ra = (r < t.rows) ? r : (t.rows - 1);   // clamp: always valid
    aptr[i]  = inp + (size_t)(t.row0 + ra) * DIN + c;
    boffv[i] = (unsigned)((r * DIN + c) * sizeof(float));
    bdst0[i] = (unsigned)(size_t)&sBf[0][r * BSTRIDE + c];
    adst0[i] = (unsigned)(r * LDSS + c);
  }
  const unsigned bBufStep = (unsigned)(BN * BSTRIDE * sizeof(float));
  const unsigned aBufStep = (unsigned)(BM * LDSS);

  // ---- prologue: stage k-tile 0 into buffer 0 ----
#pragma unroll
  for (int i = 0; i < 4; ++i) {
    async_copy_b128_gvs(bdst0[i], boffv[i], Bsad0);
    F4 a = *(const F4*)(aptr[i]);
    U2 ua; ua.x = pack2f(a.x, a.y); ua.y = pack2f(a.z, a.w);
    *(U2*)&sA[0][adst0[i]] = ua;
  }
  wait_async0();
  __syncthreads();

  v8f acc[4][2];
#pragma unroll
  for (int mi = 0; mi < 4; ++mi)
#pragma unroll
    for (int ni = 0; ni < 2; ++ni) acc[mi][ni] = {};

  F4 pa[4];

  for (int kt = 0; kt < KTILES; ++kt) {
    const int cur = kt & 1;
    const int nxt = cur ^ 1;

    // issue next k-tile: async B into other buffer, A into registers
    if (kt + 1 < KTILES) {
      const int kb = (kt + 1) * BK;                       // floats
      const unsigned long long bsad = Bsad0 + (unsigned long long)kb * sizeof(float);
      const unsigned bAdd = (unsigned)nxt * bBufStep;     // uniform
#pragma unroll
      for (int i = 0; i < 4; ++i) {
        async_copy_b128_gvs(bdst0[i] + bAdd, boffv[i], bsad);
        pa[i] = *(const F4*)(aptr[i] + kb);
      }
    }

    // ---- compute on current LDS buffer ----
    const unsigned short* baseA = &sA[cur][0];
    const unsigned*       baseB = (const unsigned*)&sBf[cur][0];

    v16bf afrag[4], bfrag[2];
#pragma unroll
    for (int mi = 0; mi < 4; ++mi) {
      // ISA A 16x32 bf16 layout: lanes 0-15 K={0..7,16..23}; 16-31 K={8..15,24..31}
      const int row = waveM + mi * 16 + l15;
      V16B v;
      v.lo = *(const U4*)(baseA + row * LDSS + lhalf * 8);
      v.hi = *(const U4*)(baseA + row * LDSS + 16 + lhalf * 8);
      afrag[mi] = __builtin_bit_cast(v16bf, v);
    }
#pragma unroll
    for (int ni = 0; ni < 2; ++ni) {
      // ISA B 32x16 bf16 layout: lanes 0-15 K={0..7,16..23}; 16-31 K={8..15,24..31}, column N=l15.
      // LDS holds f32; convert 16 f32 -> 16 bf16 with 8 v_perm_b32.
      const int nrow = waveN + ni * 16 + l15;
      const unsigned* p = baseB + nrow * BSTRIDE + lhalf * 8;
      U4 q0 = *(const U4*)(p + 0);
      U4 q1 = *(const U4*)(p + 4);
      U4 q2 = *(const U4*)(p + 16);
      U4 q3 = *(const U4*)(p + 20);
      V16B v;
      v.lo.x = pack2(q0.y, q0.x); v.lo.y = pack2(q0.w, q0.z);
      v.lo.z = pack2(q1.y, q1.x); v.lo.w = pack2(q1.w, q1.z);
      v.hi.x = pack2(q2.y, q2.x); v.hi.y = pack2(q2.w, q2.z);
      v.hi.z = pack2(q3.y, q3.x); v.hi.w = pack2(q3.w, q3.z);
      bfrag[ni] = __builtin_bit_cast(v16bf, v);
    }

#pragma unroll
    for (int mi = 0; mi < 4; ++mi)
#pragma unroll
      for (int ni = 0; ni < 2; ++ni)
        acc[mi][ni] = __builtin_amdgcn_wmma_f32_16x16x32_bf16(
            /*neg_a=*/false, afrag[mi], /*neg_b=*/false, bfrag[ni],
            /*c_mod=*/(short)0, acc[mi][ni],
            /*reuse_a=*/false, /*reuse_b=*/false);

    // ---- finish staging next buffer, then sync ----
    if (kt + 1 < KTILES) {
      const unsigned aAdd = (unsigned)nxt * aBufStep;     // uniform
#pragma unroll
      for (int i = 0; i < 4; ++i) {
        U2 ua; ua.x = pack2f(pa[i].x, pa[i].y); ua.y = pack2f(pa[i].z, pa[i].w);
        *(U2*)&sA[0][adst0[i] + aAdd] = ua;
      }
      wait_async0();       // drain this wave's async B copies into nxt
      __syncthreads();
    }
  }

  // ---- epilogue: C layout VGPR i -> (M = base + 8*lhalf + i, N = l15) ----
  const float* bv = bias + (size_t)t.e * DOUT;
  if (t.rows == BM) {
#pragma unroll
    for (int ni = 0; ni < 2; ++ni) {
      const int col = n0 + waveN + ni * 16 + l15;
      const float bcol = bv[col];
#pragma unroll
      for (int mi = 0; mi < 4; ++mi) {
        const int rbase = waveM + mi * 16 + lhalf * 8;
#pragma unroll
        for (int i = 0; i < 8; ++i)
          out[(size_t)(t.row0 + rbase + i) * DOUT + col] = acc[mi][ni][i] + bcol;
      }
    }
  } else {
#pragma unroll
    for (int ni = 0; ni < 2; ++ni) {
      const int col = n0 + waveN + ni * 16 + l15;
      const float bcol = bv[col];
#pragma unroll
      for (int mi = 0; mi < 4; ++mi) {
        const int rbase = waveM + mi * 16 + lhalf * 8;
#pragma unroll
        for (int i = 0; i < 8; ++i) {
          const int m = rbase + i;
          if (m < t.rows)
            out[(size_t)(t.row0 + m) * DOUT + col] = acc[mi][ni][i] + bcol;
        }
      }
    }
  }
}

// ---------------------------------------------------------------------------
extern "C" void kernel_launch(void* const* d_in, const int* in_sizes, int n_in,
                              void* d_out, int out_size, void* d_ws, size_t ws_size,
                              hipStream_t stream) {
  const float* inp        = (const float*)d_in[0];
  const float* weight     = (const float*)d_in[1];
  const float* bias       = (const float*)d_in[2];
  const long long* counts = (const long long*)d_in[3];   // int64 per reference
  float* out = (float*)d_out;

  int*  meta  = (int*)d_ws;
  Tile* tiles = (Tile*)((char*)d_ws + 16);

  build_tiles_kernel<<<1, 1, 0, stream>>>(counts, meta, tiles);
  moe_gemm_kernel<<<dim3(DOUT / BN, MAX_TILES), 256, 0, stream>>>(
      inp, weight, bias, meta, tiles, out);
}
